// CodeCloneDetection_67740224193145
// MI455X (gfx1250) — compile-verified, matrix operand order checked by reading, not run on hardware
//
#include <hip/hip_runtime.h>
#include <hip/hip_bf16.h>

typedef __attribute__((ext_vector_type(16))) _Float16 v16h;
typedef __attribute__((ext_vector_type(8)))  float    v8f;

#define ALPHA_LRELU 0.2f

__device__ __forceinline__ float sigm(float x) { return 1.0f / (1.0f + __expf(-x)); }
__device__ __forceinline__ float eluf(float x) { return x > 0.0f ? x : (__expf(x) - 1.0f); }

// ---------------------------------------------------------------------------
// WMMA GEMM: C[M,N] = act(A[M,K] @ B[K,N]), f32 storage, f16 math, f32 accum.
// B is supplied PRE-TRANSPOSED as Bt[N,K] (row stride ldb == K) so both A and
// B fragments are built from contiguous 8-float runs -> global_load_b128,
// no per-element 64-bit address math or load-count drains.
// One wave (32 threads) computes a 16x32 C tile: two 16x16 WMMA accumulators
// sharing one A fragment. K % 32 == 0 at all call sites; unroll 2 gives
// 4 back-to-back v_wmma per steady-state iteration.
// Batched via blockIdx.z with element strides sA/sB/sC. act: 0 = none, 1 = ELU.
// Fragment layout per CDNA5 ISA §7.12.2:
//   A (16-bit, 16x32): lane lo = M row; k = k0 + hi*8 + (p<8 ? p : p+8)
//   B mirrored with lo = N column; C/D: lane lo = N, VGPR r -> M = r + hi*8
// ---------------------------------------------------------------------------
__global__ __launch_bounds__(32) void ccd_gemm_wmma(
    const float* __restrict__ A, const float* __restrict__ Bt, float* __restrict__ C,
    int M, int N, int K, int lda, int ldb, int ldc,
    long sA, long sB, long sC, int act)
{
    A  += (long)blockIdx.z * sA;
    Bt += (long)blockIdx.z * sB;
    C  += (long)blockIdx.z * sC;

    const int tm   = blockIdx.y * 16;
    const int tn   = blockIdx.x * 32;
    const int lane = threadIdx.x;
    const int lo   = lane & 15;
    const int hi   = lane >> 4;

    const int row   = tm + lo;                      // A row for this lane
    const int col0  = tn + lo;                      // C columns for this lane
    const int col1  = tn + 16 + lo;
    const int rrow  = row  < M ? row  : (M - 1);    // clamp + zero-mask (keeps
    const int ccol0 = col0 < N ? col0 : (N - 1);    //  EXEC uniform across WMMA)
    const int ccol1 = col1 < N ? col1 : (N - 1);
    const float am  = row  < M ? 1.0f : 0.0f;
    const float bm0 = col0 < N ? 1.0f : 0.0f;
    const float bm1 = col1 < N ? 1.0f : 0.0f;
    const float* __restrict__ Ar  = A  + (long)rrow  * lda;
    const float* __restrict__ Br0 = Bt + (long)ccol0 * ldb;
    const float* __restrict__ Br1 = Bt + (long)ccol1 * ldb;

    v8f acc0 = {}, acc1 = {};
#pragma unroll 2
    for (int k0 = 0; k0 < K; k0 += 32) {
        if (k0 + 32 < K) {   // cover L2 latency of the next K panel
            __builtin_prefetch((const void*)(Ar  + k0 + 32), 0, 1);
            __builtin_prefetch((const void*)(Br0 + k0 + 32), 0, 1);
        }
        v16h af, bf0, bf1;
#pragma unroll
        for (int p = 0; p < 16; ++p) {
            const int k = k0 + hi * 8 + (p < 8 ? p : p + 8);
            af[p]  = (_Float16)(Ar[k]  * am);
            bf0[p] = (_Float16)(Br0[k] * bm0);
            bf1[p] = (_Float16)(Br1[k] * bm1);
        }
        acc0 = __builtin_amdgcn_wmma_f32_16x16x32_f16(
            false, af, false, bf0, (short)0, acc0, false, false);
        acc1 = __builtin_amdgcn_wmma_f32_16x16x32_f16(
            false, af, false, bf1, (short)0, acc1, false, false);
    }

    const bool doAct = (act == 1);
#pragma unroll
    for (int r = 0; r < 8; ++r) {
        const int m = tm + r + hi * 8;
        float v0 = acc0[r];
        float v1 = acc1[r];
        if (doAct) {                 // branchless ELU: v_exp + v_cndmask
            v0 = v0 > 0.0f ? v0 : (__expf(v0) - 1.0f);
            v1 = v1 > 0.0f ? v1 : (__expf(v1) - 1.0f);
        }
        if (m < M) {
            if (col0 < N) C[(long)m * ldc + col0] = v0;
            if (col1 < N) C[(long)m * ldc + col1] = v1;
        }
    }
}

// out[b*sOut + n*K + k] = in[b*sIn + k*N + n]   (weight pre-transpose, batched)
__global__ void ccd_transpose(const float* __restrict__ in, float* __restrict__ out,
                              int K, int N, long sIn, long sOut)
{
    const long i = (long)blockIdx.x * blockDim.x + threadIdx.x;
    if (i >= (long)K * N) return;
    const int k = (int)(i / N);
    const int n = (int)(i % N);
    out[(long)blockIdx.z * sOut + (long)n * K + k] = in[(long)blockIdx.z * sIn + i];
}

// out[b*sOb + r*sOr] = dot(A[b*sA + r*lda ...], v[b*sV ...])   (generic GEMV)
__global__ __launch_bounds__(256) void ccd_rowdot(
    const float* __restrict__ A, const float* __restrict__ v, float* __restrict__ out,
    int R, int K, int lda, long sA, long sV, long sOb, int sOr)
{
    const int b = blockIdx.y;
    const int r = blockIdx.x * blockDim.x + threadIdx.x;
    if (r >= R) return;
    const float* a  = A + b * sA + (long)r * lda;
    const float* vv = v + b * sV;
    float acc = 0.0f;
    for (int k = 0; k < K; ++k) acc += a[k] * vv[k];
    out[b * sOb + (long)r * sOr] = acc;
}

// GAT attention row: one block per (node n, head). Computes masked
// leaky-relu logits, row softmax, h2[n,:] = att @ Wh, optional ELU.
__global__ __launch_bounds__(256) void ccd_gat_attn(
    const float* __restrict__ s, long sS, const float* __restrict__ d, long sD,
    const float* __restrict__ Wh, long sWh, int ldWh,
    const float* __restrict__ ee, long eeStride, long eeHead,
    const float* __restrict__ adj, float* __restrict__ out,
    int ldOut, int colStride, int Nn, int ncols, int doElu)
{
    __shared__ float lg[256];
    __shared__ float red[256];
    __shared__ float sc[2];

    const int n = blockIdx.x;
    const int h = blockIdx.y;
    const int t = threadIdx.x;
    const float* sp  = s  + h * sS;
    const float* dp  = d  + h * sD;
    const float* WhH = Wh + h * sWh;
    const long   eOf = h * eeHead;

    if (t < Nn) {
        float l = sp[n] + dp[t] + ee[((long)n * Nn + t) * eeStride + eOf];
        l = l > 0.0f ? l : ALPHA_LRELU * l;
        if (!(adj[(long)n * Nn + t] > 0.0f)) l = -1e9f;
        lg[t] = l;
    }
    red[t] = (t < Nn) ? lg[t] : -3.0e38f;
    __syncthreads();
    for (int off = 128; off > 0; off >>= 1) {
        if (t < off) red[t] = fmaxf(red[t], red[t + off]);
        __syncthreads();
    }
    if (t == 0) sc[0] = red[0];
    __syncthreads();
    const float mx = sc[0];
    float ev = 0.0f;
    if (t < Nn) { ev = __expf(lg[t] - mx); lg[t] = ev; }
    red[t] = (t < Nn) ? ev : 0.0f;
    __syncthreads();
    for (int off = 128; off > 0; off >>= 1) {
        if (t < off) red[t] += red[t + off];
        __syncthreads();
    }
    if (t == 0) sc[1] = red[0];
    __syncthreads();
    const float denom = sc[1];

    for (int j = t; j < ncols; j += 256) {
        float acc = 0.0f;
        for (int m = 0; m < Nn; ++m) acc += lg[m] * WhH[(long)m * ldWh + j];
        acc /= denom;
        if (doElu) acc = eluf(acc);
        out[(long)n * ldOut + h * colStride + j] = acc;
    }
}

// GlobalAttention pool: gate = softmax(sigmoid(x@w+b), axis=0); out = sum gate*x
__global__ __launch_bounds__(256) void ccd_gpool(
    const float* __restrict__ x, int Nn, int D,
    const float* __restrict__ w, const float* __restrict__ b, float* __restrict__ out)
{
    __shared__ float g[256];
    __shared__ float red[256];
    __shared__ float sc[2];
    const int t = threadIdx.x;
    float sv = 0.0f;
    if (t < Nn) {
        float acc = b[0];
        for (int k = 0; k < D; ++k) acc += x[(long)t * D + k] * w[k];
        sv = sigm(acc);
    }
    g[t]   = sv;
    red[t] = (t < Nn) ? sv : -3.0e38f;
    __syncthreads();
    for (int off = 128; off > 0; off >>= 1) {
        if (t < off) red[t] = fmaxf(red[t], red[t + off]);
        __syncthreads();
    }
    if (t == 0) sc[0] = red[0];
    __syncthreads();
    float ev = 0.0f;
    if (t < Nn) { ev = __expf(g[t] - sc[0]); g[t] = ev; }
    red[t] = (t < Nn) ? ev : 0.0f;
    __syncthreads();
    for (int off = 128; off > 0; off >>= 1) {
        if (t < off) red[t] += red[t + off];
        __syncthreads();
    }
    if (t == 0) sc[1] = red[0];
    __syncthreads();
    const float denom = sc[1];
    for (int dd = t; dd < D; dd += 256) {
        float acc = 0.0f;
        for (int m = 0; m < Nn; ++m) acc += g[m] * x[(long)m * D + dd];
        out[dd] = acc / denom;
    }
}

// Edge pool scoring: s[e] = sigmoid([x[src],x[dst],e] . w + b); segment sums.
__global__ __launch_bounds__(128) void ccd_edge_score(
    const float* __restrict__ x, const float* __restrict__ e, const int* __restrict__ eidx,
    int E, int Dx, int De, const float* __restrict__ w, const float* __restrict__ b,
    float* __restrict__ s, float* __restrict__ deg, float* __restrict__ ssum)
{
    const int ed = blockIdx.x * blockDim.x + threadIdx.x;
    if (ed >= E) return;
    const int src = eidx[ed];
    const int dst = eidx[E + ed];
    const float* xs = x + (long)src * Dx;
    const float* xd = x + (long)dst * Dx;
    const float* ef = e + (long)ed * De;
    float acc = b[0];
    for (int k = 0; k < Dx; ++k) acc += xs[k] * w[k];
    for (int k = 0; k < Dx; ++k) acc += xd[k] * w[Dx + k];
    for (int k = 0; k < De; ++k) acc += ef[k] * w[2 * Dx + k];
    const float sv = sigm(acc);
    s[ed] = sv;
    atomicAdd(&deg[src], 1.0f);
    atomicAdd(&ssum[src], sv);
}

__global__ void ccd_scale_nodes(float* __restrict__ x, long total, int D,
                                const float* __restrict__ ssum, const float* __restrict__ deg)
{
    const long i = (long)blockIdx.x * blockDim.x + threadIdx.x;
    if (i >= total) return;
    const int n = (int)(i / D);
    x[i] *= ssum[n] / (deg[n] + 1e-6f);
}

__global__ void ccd_scale_edges(float* __restrict__ e, long total, int D,
                                const float* __restrict__ s)
{
    const long i = (long)blockIdx.x * blockDim.x + threadIdx.x;
    if (i >= total) return;
    e[i] *= s[(int)(i / D)];
}

// Layer-2 edge term: ee2[src*Nn+dst] = e[ed] . v2   (sparse n2n2 folded)
__global__ __launch_bounds__(128) void ccd_ee2_scatter(
    const float* __restrict__ e, const int* __restrict__ eidx,
    const float* __restrict__ v2, float* __restrict__ ee2, int E, int Nn, int D)
{
    const int ed = blockIdx.x * blockDim.x + threadIdx.x;
    if (ed >= E) return;
    const int src = eidx[ed];
    const int dst = eidx[E + ed];
    float acc = 0.0f;
    const float* ef = e + (long)ed * D;
    for (int k = 0; k < D; ++k) acc += ef[k] * v2[k];
    ee2[(long)src * Nn + dst] = acc;
}

__global__ void ccd_memzero(float* __restrict__ p, long n)
{
    const long i = (long)blockIdx.x * blockDim.x + threadIdx.x;
    if (i < n) p[i] = 0.0f;
}

// Bidirectional 2-layer LSTM over T=2 + FC + softmax, single block of 512.
__global__ __launch_bounds__(512) void ccd_lstm_head(
    const float* __restrict__ xseq,
    const float* __restrict__ Wih0, const float* __restrict__ Whh0, const float* __restrict__ b0,
    const float* __restrict__ Wih1, const float* __restrict__ Whh1, const float* __restrict__ b1,
    const float* __restrict__ fcw, const float* __restrict__ fcb, float* __restrict__ out)
{
    __shared__ float xb[2 * 2240];
    __shared__ float y0[2 * 256];
    __shared__ float y1[2 * 256];
    __shared__ float hs[128], cs[128];
    __shared__ float gates[512];
    const int t = threadIdx.x;
    for (int i = t; i < 2 * 2240; i += 512) xb[i] = xseq[i];
    __syncthreads();

    // layer 0: Din = 2240
    for (int dir = 0; dir < 2; ++dir) {
        const float* Wih = Wih0 + (long)dir * 512 * 2240;
        const float* Whh = Whh0 + (long)dir * 512 * 128;
        const float* bb  = b0 + dir * 512;
        if (t < 128) { hs[t] = 0.0f; cs[t] = 0.0f; }
        __syncthreads();
        for (int st = 0; st < 2; ++st) {
            const int tt = dir ? 1 - st : st;
            float acc = bb[t];
            const float* xr = xb + tt * 2240;
            const float* wr = Wih + (long)t * 2240;
            for (int k = 0; k < 2240; ++k) acc += wr[k] * xr[k];
            const float* hr = Whh + (long)t * 128;
            for (int k = 0; k < 128; ++k) acc += hr[k] * hs[k];
            gates[t] = acc;
            __syncthreads();
            if (t < 128) {
                const float iv = sigm(gates[t]);
                const float fv = sigm(gates[128 + t]);
                const float gv = tanhf(gates[256 + t]);
                const float ov = sigm(gates[384 + t]);
                const float cn = fv * cs[t] + iv * gv;
                cs[t] = cn;
                const float hn = ov * tanhf(cn);
                hs[t] = hn;
                y0[tt * 256 + dir * 128 + t] = hn;
            }
            __syncthreads();
        }
    }
    // layer 1: Din = 256
    for (int dir = 0; dir < 2; ++dir) {
        const float* Wih = Wih1 + (long)dir * 512 * 256;
        const float* Whh = Whh1 + (long)dir * 512 * 128;
        const float* bb  = b1 + dir * 512;
        if (t < 128) { hs[t] = 0.0f; cs[t] = 0.0f; }
        __syncthreads();
        for (int st = 0; st < 2; ++st) {
            const int tt = dir ? 1 - st : st;
            float acc = bb[t];
            const float* xr = y0 + tt * 256;
            const float* wr = Wih + (long)t * 256;
            for (int k = 0; k < 256; ++k) acc += wr[k] * xr[k];
            const float* hr = Whh + (long)t * 128;
            for (int k = 0; k < 128; ++k) acc += hr[k] * hs[k];
            gates[t] = acc;
            __syncthreads();
            if (t < 128) {
                const float iv = sigm(gates[t]);
                const float fv = sigm(gates[128 + t]);
                const float gv = tanhf(gates[256 + t]);
                const float ov = sigm(gates[384 + t]);
                const float cn = fv * cs[t] + iv * gv;
                cs[t] = cn;
                const float hn = ov * tanhf(cn);
                hs[t] = hn;
                y1[tt * 256 + dir * 128 + t] = hn;
            }
            __syncthreads();
        }
    }
    if (t < 2) {
        float acc = fcb[t];
        for (int k = 0; k < 256; ++k) acc += y1[256 + k] * fcw[k * 2 + t];
        gates[t] = acc;
    }
    __syncthreads();
    if (t == 0) {
        const float m  = fmaxf(gates[0], gates[1]);
        const float e0 = __expf(gates[0] - m);
        const float e1 = __expf(gates[1] - m);
        out[0] = e0 / (e0 + e1);
        out[1] = e1 / (e0 + e1);
    }
}

// ---------------------------------------------------------------------------
extern "C" void kernel_launch(void* const* d_in, const int* in_sizes, int n_in,
                              void* d_out, int out_size, void* d_ws, size_t ws_size,
                              hipStream_t stream)
{
    (void)in_sizes; (void)n_in; (void)out_size; (void)ws_size;
    const int NN = 200, EE = 3200, HID = 64, NH = 16, OUT = 128, CAT = 2048;

    const float* W_h    = (const float*)d_in[10];
    const float* W_gat  = (const float*)d_in[11];
    const float* a_src  = (const float*)d_in[12];
    const float* a_dst  = (const float*)d_in[13];
    const float* a_e    = (const float*)d_in[14];
    const float* We_gat = (const float*)d_in[15];
    const float* W_out  = (const float*)d_in[16];
    const float* ao_src = (const float*)d_in[17];
    const float* ao_dst = (const float*)d_in[18];
    const float* ao_e   = (const float*)d_in[19];
    const float* We_out = (const float*)d_in[20];
    const float* ep1_w  = (const float*)d_in[21];
    const float* ep1_b  = (const float*)d_in[22];
    const float* ep2_w  = (const float*)d_in[23];
    const float* ep2_b  = (const float*)d_in[24];
    const float* g1_w   = (const float*)d_in[25];
    const float* g1_b   = (const float*)d_in[26];
    const float* g2_w   = (const float*)d_in[27];
    const float* g2_b   = (const float*)d_in[28];
    const float* g3_w   = (const float*)d_in[29];
    const float* g3_b   = (const float*)d_in[30];
    const float* Wih0   = (const float*)d_in[31];
    const float* Whh0   = (const float*)d_in[32];
    const float* b0     = (const float*)d_in[33];
    const float* Wih1   = (const float*)d_in[34];
    const float* Whh1   = (const float*)d_in[35];
    const float* b1     = (const float*)d_in[36];
    const float* fc_w   = (const float*)d_in[37];
    const float* fc_b   = (const float*)d_in[38];

    // workspace layout (floats)
    float* ws   = (float*)d_ws;
    float* h0   = ws;  ws += NN * HID;            // [200,64]
    float* VEt  = ws;  ws += NH * HID;            // [16,64]  VE pre-transposed
    float* WhA  = ws;  ws += (long)NH * NN * OUT; // [16][200,128]
    float* sAr  = ws;  ws += NH * NN;
    float* dAr  = ws;  ws += NH * NN;
    float* EE1  = ws;  ws += (long)NN * NN * NH;  // [40000,16]
    float* hcat = ws;  ws += (long)NN * CAT;      // [200,2048]
    float* ecat = ws;  ws += (long)EE * CAT;      // [3200,2048]
    float* deg  = ws;  ws += 256;
    float* ssum = ws;  ws += 256;
    float* sedg = ws;  ws += EE;
    float* v2   = ws;  ws += CAT;
    float* Wh2  = ws;  ws += NN * OUT;
    float* s2   = ws;  ws += 256;
    float* d2v  = ws;  ws += 256;
    float* ee2  = ws;  ws += NN * NN;
    float* h3   = ws;  ws += NN * OUT;
    float* e3   = ws;  ws += (long)EE * OUT;
    float* xseq = ws;  ws += 2 * 2240;
    // pre-transposed weights ([N,K] layout for WMMA B fragments)
    float* Wt_h    = ws;  ws += HID * HID;            // [64,64]
    float* WtA     = ws;  ws += (long)NH * OUT * HID; // [16][128,64]
    float* WtE     = ws;  ws += (long)NH * OUT * HID; // [16][128,64]
    float* Wt_out  = ws;  ws += (long)OUT * CAT;      // [128,2048]
    float* Wet_out = ws;  ws += (long)OUT * CAT;      // [128,2048]

    // one-time weight transposes (tiny: ~0.8M elements total, L2 resident)
    ccd_transpose<<<dim3((HID * HID + 255) / 256, 1, 1), 256, 0, stream>>>(
        W_h, Wt_h, HID, HID, 0, 0);
    ccd_transpose<<<dim3((HID * OUT + 255) / 256, 1, NH), 256, 0, stream>>>(
        W_gat, WtA, HID, OUT, (long)HID * OUT, (long)OUT * HID);
    ccd_transpose<<<dim3((HID * OUT + 255) / 256, 1, NH), 256, 0, stream>>>(
        We_gat, WtE, HID, OUT, (long)HID * OUT, (long)OUT * HID);
    ccd_transpose<<<dim3(((long)CAT * OUT + 255) / 256, 1, 1), 256, 0, stream>>>(
        W_out, Wt_out, CAT, OUT, 0, 0);
    ccd_transpose<<<dim3(((long)CAT * OUT + 255) / 256, 1, 1), 256, 0, stream>>>(
        We_out, Wet_out, CAT, OUT, 0, 0);

    // v2 = We_out @ ao_e  (layer-2 edge vector, graph-independent)
    ccd_rowdot<<<dim3(8, 1), 256, 0, stream>>>(We_out, ao_e, v2, CAT, OUT, OUT, 0, 0, 0, 1);
    // VEt[i,k] = sum_j We_gat[i][k][j] * a_e[i][j]  (emitted directly transposed)
    ccd_rowdot<<<dim3(1, NH), 256, 0, stream>>>(We_gat, a_e, VEt, HID, OUT, OUT,
                                                (long)HID * OUT, OUT, HID, 1);

    for (int g = 0; g < 2; ++g) {
        const float* feat  = (const float*)d_in[0 + g * 5];
        const int*   eidx  = (const int*)  d_in[1 + g * 5];
        const float* eattr = (const float*)d_in[2 + g * 5];
        const float* adj   = (const float*)d_in[3 + g * 5];
        const float* n2n   = (const float*)d_in[4 + g * 5];
        float* hsg = xseq + g * 2240;

        // h0 = elu(feat @ W_h)
        ccd_gemm_wmma<<<dim3(2, 13, 1), 32, 0, stream>>>(
            feat, Wt_h, h0, NN, HID, HID, HID, HID, HID, 0, 0, 0, 1);
        ccd_gpool<<<1, 256, 0, stream>>>(h0, NN, HID, g1_w, g1_b, hsg);

        // per-head Wh, attention vectors, edge term
        ccd_gemm_wmma<<<dim3(4, 13, NH), 32, 0, stream>>>(
            h0, WtA, WhA, NN, OUT, HID, HID, HID, OUT,
            0, (long)OUT * HID, (long)NN * OUT, 0);
        ccd_rowdot<<<dim3(1, NH), 256, 0, stream>>>(WhA, a_src, sAr, NN, OUT, OUT,
                                                    (long)NN * OUT, OUT, NN, 1);
        ccd_rowdot<<<dim3(1, NH), 256, 0, stream>>>(WhA, a_dst, dAr, NN, OUT, OUT,
                                                    (long)NN * OUT, OUT, NN, 1);
        // EE1 = n2n @ VE   ([40000,64]x[64,16], folded sparse edge term)
        ccd_gemm_wmma<<<dim3(1, 2500, 1), 32, 0, stream>>>(
            n2n, VEt, EE1, NN * NN, NH, HID, HID, HID, NH, 0, 0, 0, 0);
        // ecat[:, h*128:(h+1)*128] = elu(eattr @ We_gat[h])
        ccd_gemm_wmma<<<dim3(4, 200, NH), 32, 0, stream>>>(
            eattr, WtE, ecat, EE, OUT, HID, HID, HID, CAT,
            0, (long)OUT * HID, OUT, 1);
        // layer-1 attention -> hcat (elu)
        ccd_gat_attn<<<dim3(NN, NH), 256, 0, stream>>>(
            sAr, NN, dAr, NN, WhA, (long)NN * OUT, OUT, EE1, NH, 1,
            adj, hcat, CAT, OUT, NN, OUT, 1);

        // edge pool 1
        ccd_memzero<<<2, 256, 0, stream>>>(deg, 512);
        ccd_edge_score<<<25, 128, 0, stream>>>(hcat, ecat, eidx, EE, CAT, CAT,
                                               ep1_w, ep1_b, sedg, deg, ssum);
        ccd_scale_nodes<<<(NN * CAT + 255) / 256, 256, 0, stream>>>(
            hcat, (long)NN * CAT, CAT, ssum, deg);
        ccd_scale_edges<<<((long)EE * CAT + 255) / 256, 256, 0, stream>>>(
            ecat, (long)EE * CAT, CAT, sedg);
        ccd_gpool<<<1, 256, 0, stream>>>(hcat, NN, CAT, g2_w, g2_b, hsg + 64);

        // layer 2 GAT
        ccd_gemm_wmma<<<dim3(4, 13, 1), 32, 0, stream>>>(
            hcat, Wt_out, Wh2, NN, OUT, CAT, CAT, CAT, OUT, 0, 0, 0, 0);
        ccd_rowdot<<<dim3(1, 1), 256, 0, stream>>>(Wh2, ao_src, s2, NN, OUT, OUT, 0, 0, 0, 1);
        ccd_rowdot<<<dim3(1, 1), 256, 0, stream>>>(Wh2, ao_dst, d2v, NN, OUT, OUT, 0, 0, 0, 1);
        ccd_memzero<<<(NN * NN + 255) / 256, 256, 0, stream>>>(ee2, NN * NN);
        ccd_ee2_scatter<<<25, 128, 0, stream>>>(ecat, eidx, v2, ee2, EE, NN, CAT);
        ccd_gat_attn<<<dim3(NN, 1), 256, 0, stream>>>(
            s2, 0, d2v, 0, Wh2, 0, OUT, ee2, 1, 0,
            adj, h3, OUT, 0, NN, OUT, 0);
        // e3 = elu(ecat @ We_out)   -- the dominant GEMM (K = 2048)
        ccd_gemm_wmma<<<dim3(4, 200, 1), 32, 0, stream>>>(
            ecat, Wet_out, e3, EE, OUT, CAT, CAT, CAT, OUT, 0, 0, 0, 1);

        // edge pool 2 (scaled e3 unused afterwards -> skip edge scaling)
        ccd_memzero<<<2, 256, 0, stream>>>(deg, 512);
        ccd_edge_score<<<25, 128, 0, stream>>>(h3, e3, eidx, EE, OUT, OUT,
                                               ep2_w, ep2_b, sedg, deg, ssum);
        ccd_scale_nodes<<<(NN * OUT + 255) / 256, 256, 0, stream>>>(
            h3, (long)NN * OUT, OUT, ssum, deg);
        ccd_gpool<<<1, 256, 0, stream>>>(h3, NN, OUT, g3_w, g3_b, hsg + 64 + CAT);
    }

    ccd_lstm_head<<<1, 512, 0, stream>>>(xseq, Wih0, Whh0, b0, Wih1, Whh1, b1,
                                         fc_w, fc_b, (float*)d_out);
}
